// TernaryLinear_75874892251910
// MI455X (gfx1250) — matrix operand order repaired and checked by reading, not compiled, not run on hardware
//
#include <hip/hip_runtime.h>
#include <stdint.h>

typedef __attribute__((ext_vector_type(16))) __bf16 v16bf;
typedef __attribute__((ext_vector_type(8)))  float  v8f;

#define BM   128
#define BN   128
#define BK   32
#define LDT  40          // bf16-GEMM LDS row stride (80B, 16B-aligned for async b128)
#define FLDA 36          // fallback-kernel LDS row stride
#define KDIM 4096
#define NDIM 4096

#define WS_Q_OFF   0ULL
#define WS_XH_OFF  33554432ULL     // 4096*4096*2
#define WS_XL_OFF  100663296ULL    // + 8192*4096*2
#define WS_NEEDED  167772160ULL    // + 8192*4096*2

union FragU {
  v16bf v;
  uint2 u[4];
};

__device__ __forceinline__ unsigned short f32_to_bf16_rne(float f) {
  unsigned int u = __float_as_uint(f);
  unsigned int r = u + 0x7FFFu + ((u >> 16) & 1u);
  return (unsigned short)(r >> 16);
}
__device__ __forceinline__ float bf16_to_f32(unsigned short h) {
  return __uint_as_float(((unsigned int)h) << 16);
}
__device__ __forceinline__ unsigned short ternary_q(float wv) {
  return (fabsf(wv) > 0.1f) ? (unsigned short)((wv > 0.0f) ? 0x3F80u : 0xBF80u)
                            : (unsigned short)0u;
}

// ---------------------------------------------------------------------------
// Prep pass 1: weight f32 -> ternary bf16 {-1,0,+1}  (memory bound)
// ---------------------------------------------------------------------------
__global__ __launch_bounds__(256)
void prep_quant(const float* __restrict__ w, unsigned short* __restrict__ q) {
  const size_t i = ((size_t)blockIdx.x * 256 + threadIdx.x) * 8;
  const float4* s = (const float4*)(w + i);
  const float4 a = s[0], b = s[1];
  const float v[8] = {a.x, a.y, a.z, a.w, b.x, b.y, b.z, b.w};
  unsigned short t[8];
#pragma unroll
  for (int e = 0; e < 8; ++e) t[e] = ternary_q(v[e]);
  uint4 o;
  o.x = (unsigned)t[0] | ((unsigned)t[1] << 16);
  o.y = (unsigned)t[2] | ((unsigned)t[3] << 16);
  o.z = (unsigned)t[4] | ((unsigned)t[5] << 16);
  o.w = (unsigned)t[6] | ((unsigned)t[7] << 16);
  *(uint4*)(q + i) = o;
}

// ---------------------------------------------------------------------------
// Prep pass 2: x f32 -> bf16x2 split (hi = bf16(x), lo = bf16(x - hi))
// ---------------------------------------------------------------------------
__global__ __launch_bounds__(256)
void prep_split(const float* __restrict__ x,
                unsigned short* __restrict__ xh,
                unsigned short* __restrict__ xl) {
  const size_t i = ((size_t)blockIdx.x * 256 + threadIdx.x) * 8;
  const float4* s = (const float4*)(x + i);
  const float4 a = s[0], b = s[1];
  const float v[8] = {a.x, a.y, a.z, a.w, b.x, b.y, b.z, b.w};
  unsigned short th[8], tl[8];
#pragma unroll
  for (int e = 0; e < 8; ++e) {
    const unsigned short h = f32_to_bf16_rne(v[e]);
    th[e] = h;
    tl[e] = f32_to_bf16_rne(v[e] - bf16_to_f32(h));
  }
  uint4 oh, ol;
  oh.x = (unsigned)th[0] | ((unsigned)th[1] << 16);
  oh.y = (unsigned)th[2] | ((unsigned)th[3] << 16);
  oh.z = (unsigned)th[4] | ((unsigned)th[5] << 16);
  oh.w = (unsigned)th[6] | ((unsigned)th[7] << 16);
  ol.x = (unsigned)tl[0] | ((unsigned)tl[1] << 16);
  ol.y = (unsigned)tl[2] | ((unsigned)tl[3] << 16);
  ol.z = (unsigned)tl[4] | ((unsigned)tl[5] << 16);
  ol.w = (unsigned)tl[6] | ((unsigned)tl[7] << 16);
  *(uint4*)(xh + i) = oh;
  *(uint4*)(xl + i) = ol;
}

// ---------------------------------------------------------------------------
// Main GEMM: pre-converted bf16 operands, async-to-LDS staging, WMMA compute.
// Block = 256 threads (8 waves), tile 128x128, BK=32, double-buffered LDS.
// Waves: 4(M) x 2(N); wave tile 32x64 -> mt=2, nt=4 -> 16 WMMA / wave / k-step.
// ---------------------------------------------------------------------------
__global__ __launch_bounds__(256)
void gemm_bf16x2_wmma(const unsigned short* __restrict__ xh,
                      const unsigned short* __restrict__ xl,
                      const unsigned short* __restrict__ qw,
                      const float* __restrict__ bias,
                      float* __restrict__ out) {
  __shared__ unsigned short As_hi[2][BM * LDT];
  __shared__ unsigned short As_lo[2][BM * LDT];
  __shared__ unsigned short Bs  [2][BN * LDT];

  const int tid   = threadIdx.x;
  const int lane  = tid & 31;
  const int wave  = tid >> 5;
  const int waveM = wave >> 1;       // 0..3 -> 32 rows each
  const int waveN = wave & 1;        // 0..1 -> 64 cols each

  const int m0 = blockIdx.y * BM;
  const int n0 = blockIdx.x * BN;

  // ---- async tile copy mapping: thread owns 32B (16 bf16) of each 128x32 tile
  const int crow = tid >> 1;              // 0..127
  const int ccol = (tid & 1) << 4;        // 0 or 16 elements

  const uint64_t gxh = (uint64_t)(uintptr_t)(xh + (size_t)(m0 + crow) * KDIM + ccol);
  const uint64_t gxl = (uint64_t)(uintptr_t)(xl + (size_t)(m0 + crow) * KDIM + ccol);
  const uint64_t gqw = (uint64_t)(uintptr_t)(qw + (size_t)(n0 + crow) * KDIM + ccol);
  const uint32_t dh0 = (uint32_t)(uintptr_t)&As_hi[0][crow * LDT + ccol];
  const uint32_t dl0 = (uint32_t)(uintptr_t)&As_lo[0][crow * LDT + ccol];
  const uint32_t dq0 = (uint32_t)(uintptr_t)&Bs  [0][crow * LDT + ccol];
  const uint32_t bufbytes = (uint32_t)(BM * LDT * 2);

  auto copy_tile = [&](int kt, int buf) {
    const uint64_t off = (uint64_t)kt * (BK * 2);           // bytes along K
    const uint32_t dof = buf ? bufbytes : 0u;
    asm volatile("global_load_async_to_lds_b128 %0, %1, off\n\t"
                 "global_load_async_to_lds_b128 %0, %1, off offset:16"
                 :: "v"(dh0 + dof), "v"(gxh + off) : "memory");
    asm volatile("global_load_async_to_lds_b128 %0, %1, off\n\t"
                 "global_load_async_to_lds_b128 %0, %1, off offset:16"
                 :: "v"(dl0 + dof), "v"(gxl + off) : "memory");
    asm volatile("global_load_async_to_lds_b128 %0, %1, off\n\t"
                 "global_load_async_to_lds_b128 %0, %1, off offset:16"
                 :: "v"(dq0 + dof), "v"(gqw + off) : "memory");
  };

  // ---- WMMA fragment lane mapping (per ISA 05_wmma VGPR layouts)
  const int am  = waveM * 32;
  const int bn  = waveN * 64;
  const int l15 = lane & 15;
  const int kbA = (lane < 16) ? 0 : 8;    // A: VGPR0-3 -> K kb..kb+7, VGPR4-7 -> K kb+16..kb+23
  const int kbB = (lane < 16) ? 0 : 8;    // B: VGPR0-3 -> K kb..kb+7, VGPR4-7 -> K kb+16..kb+23

  v8f acc[2][4];
#pragma unroll
  for (int mt = 0; mt < 2; ++mt)
#pragma unroll
    for (int nt = 0; nt < 4; ++nt)
      acc[mt][nt] = (v8f){0.f, 0.f, 0.f, 0.f, 0.f, 0.f, 0.f, 0.f};

  auto compute = [&](int buf) {
    FragU bfr[4];
#pragma unroll
    for (int nt = 0; nt < 4; ++nt) {
      const unsigned short* bb = &Bs[buf][(bn + nt * 16 + l15) * LDT + kbB];
      bfr[nt].u[0] = *(const uint2*)(bb + 0);
      bfr[nt].u[1] = *(const uint2*)(bb + 4);
      bfr[nt].u[2] = *(const uint2*)(bb + 16);
      bfr[nt].u[3] = *(const uint2*)(bb + 20);
    }
#pragma unroll
    for (int mt = 0; mt < 2; ++mt) {
      const unsigned short* ah = &As_hi[buf][(am + mt * 16 + l15) * LDT + kbA];
      const unsigned short* al = &As_lo[buf][(am + mt * 16 + l15) * LDT + kbA];
      FragU fh, fl;
      fh.u[0] = *(const uint2*)(ah + 0);
      fh.u[1] = *(const uint2*)(ah + 4);
      fh.u[2] = *(const uint2*)(ah + 16);
      fh.u[3] = *(const uint2*)(ah + 20);
      fl.u[0] = *(const uint2*)(al + 0);
      fl.u[1] = *(const uint2*)(al + 4);
      fl.u[2] = *(const uint2*)(al + 16);
      fl.u[3] = *(const uint2*)(al + 20);
#pragma unroll
      for (int nt = 0; nt < 4; ++nt) {
        acc[mt][nt] = __builtin_amdgcn_wmma_f32_16x16x32_bf16(
            false, fh.v, false, bfr[nt].v, (short)0, acc[mt][nt], false, false);
        acc[mt][nt] = __builtin_amdgcn_wmma_f32_16x16x32_bf16(
            false, fl.v, false, bfr[nt].v, (short)0, acc[mt][nt], false, false);
      }
    }
  };

  // ---- pipelined main loop: async copy of tile kt+1 overlaps WMMAs on tile kt
  copy_tile(0, 0);
  asm volatile("s_wait_asynccnt 0x0" ::: "memory");
  __syncthreads();

  const int KT = KDIM / BK;  // 128
#pragma unroll 1
  for (int kt = 0; kt < KT; ++kt) {
    const int cur = kt & 1;
    if (kt + 1 < KT) copy_tile(kt + 1, cur ^ 1);
    compute(cur);
    asm volatile("s_wait_asynccnt 0x0" ::: "memory");
    __syncthreads();
  }

  // ---- epilogue: C layout -> VGPR r holds M = r + 8*(lane>=16), N = lane&15
  const int mhi = (lane >> 4) << 3;
#pragma unroll
  for (int nt = 0; nt < 4; ++nt) {
    const int n  = n0 + bn + nt * 16 + l15;
    const float bv = bias[n];
#pragma unroll
    for (int mt = 0; mt < 2; ++mt) {
#pragma unroll
      for (int r = 0; r < 8; ++r) {
        const int m = m0 + am + mt * 16 + mhi + r;
        out[(size_t)m * NDIM + n] = acc[mt][nt][r] + bv;
      }
    }
  }
}

// ---------------------------------------------------------------------------
// Fallback (ws too small): round-1 fused kernel, conversions in the k-loop.
// ---------------------------------------------------------------------------
__global__ __launch_bounds__(256)
void ternary_linear_fused(const float* __restrict__ x,
                          const float* __restrict__ w,
                          const float* __restrict__ bias,
                          float* __restrict__ out) {
  __shared__ unsigned short As_hi[2][BM * FLDA];
  __shared__ unsigned short As_lo[2][BM * FLDA];
  __shared__ unsigned short Bs  [2][BN * FLDA];

  const int tid   = threadIdx.x;
  const int lane  = tid & 31;
  const int wave  = tid >> 5;
  const int waveM = wave >> 2;
  const int waveN = wave & 3;

  const int m0 = blockIdx.y * BM;
  const int n0 = blockIdx.x * BN;

  const int lrow = tid >> 1;
  const int lcol = (tid & 1) << 4;

  const float* xg = x + (size_t)(m0 + lrow) * KDIM + lcol;
  const float* wg = w + (size_t)(n0 + lrow) * KDIM + lcol;

  const int am  = waveM * 64;
  const int bn  = waveN * 32;
  const int l15 = lane & 15;
  const int kbA = (lane < 16) ? 0 : 8;
  const int kbB = (lane < 16) ? 0 : 8;

  v8f acc[4][2];
#pragma unroll
  for (int mt = 0; mt < 4; ++mt)
#pragma unroll
    for (int nt = 0; nt < 2; ++nt)
      acc[mt][nt] = (v8f){0.f, 0.f, 0.f, 0.f, 0.f, 0.f, 0.f, 0.f};

  float4 xa[4], wa[4];

  auto fetch = [&](int kt) {
    const float4* xp = (const float4*)(xg + kt * BK);
    const float4* wp = (const float4*)(wg + kt * BK);
#pragma unroll
    for (int j = 0; j < 4; ++j) { xa[j] = xp[j]; wa[j] = wp[j]; }
  };

  auto stage = [&](int buf) {
    unsigned short hh[16], ll[16], qq[16];
#pragma unroll
    for (int j = 0; j < 4; ++j) {
      const float xv[4] = {xa[j].x, xa[j].y, xa[j].z, xa[j].w};
      const float wv[4] = {wa[j].x, wa[j].y, wa[j].z, wa[j].w};
#pragma unroll
      for (int e = 0; e < 4; ++e) {
        const unsigned short h = f32_to_bf16_rne(xv[e]);
        hh[4 * j + e] = h;
        ll[4 * j + e] = f32_to_bf16_rne(xv[e] - bf16_to_f32(h));
        qq[4 * j + e] = ternary_q(wv[e]);
      }
    }
    const int ab = lrow * FLDA + lcol;
#pragma unroll
    for (int j = 0; j < 4; ++j) {
      uint2 th, tl, tq;
      th.x = (unsigned)hh[4 * j + 0] | ((unsigned)hh[4 * j + 1] << 16);
      th.y = (unsigned)hh[4 * j + 2] | ((unsigned)hh[4 * j + 3] << 16);
      tl.x = (unsigned)ll[4 * j + 0] | ((unsigned)ll[4 * j + 1] << 16);
      tl.y = (unsigned)ll[4 * j + 2] | ((unsigned)ll[4 * j + 3] << 16);
      tq.x = (unsigned)qq[4 * j + 0] | ((unsigned)qq[4 * j + 1] << 16);
      tq.y = (unsigned)qq[4 * j + 2] | ((unsigned)qq[4 * j + 3] << 16);
      *(uint2*)&As_hi[buf][ab + 4 * j] = th;
      *(uint2*)&As_lo[buf][ab + 4 * j] = tl;
      *(uint2*)&Bs  [buf][ab + 4 * j] = tq;
    }
  };

  auto compute = [&](int buf) {
    FragU bfr[2];
#pragma unroll
    for (int nt = 0; nt < 2; ++nt) {
      const unsigned short* bb = &Bs[buf][(bn + nt * 16 + l15) * FLDA + kbB];
      bfr[nt].u[0] = *(const uint2*)(bb + 0);
      bfr[nt].u[1] = *(const uint2*)(bb + 4);
      bfr[nt].u[2] = *(const uint2*)(bb + 16);
      bfr[nt].u[3] = *(const uint2*)(bb + 20);
    }
#pragma unroll
    for (int mt = 0; mt < 4; ++mt) {
      const unsigned short* ah = &As_hi[buf][(am + mt * 16 + l15) * FLDA + kbA];
      const unsigned short* al = &As_lo[buf][(am + mt * 16 + l15) * FLDA + kbA];
      FragU fh, fl;
      fh.u[0] = *(const uint2*)(ah + 0);
      fh.u[1] = *(const uint2*)(ah + 4);
      fh.u[2] = *(const uint2*)(ah + 16);
      fh.u[3] = *(const uint2*)(ah + 20);
      fl.u[0] = *(const uint2*)(al + 0);
      fl.u[1] = *(const uint2*)(al + 4);
      fl.u[2] = *(const uint2*)(al + 16);
      fl.u[3] = *(const uint2*)(al + 20);
#pragma unroll
      for (int nt = 0; nt < 2; ++nt) {
        acc[mt][nt] = __builtin_amdgcn_wmma_f32_16x16x32_bf16(
            false, fh.v, false, bfr[nt].v, (short)0, acc[mt][nt], false, false);
        acc[mt][nt] = __builtin_amdgcn_wmma_f32_16x16x32_bf16(
            false, fl.v, false, bfr[nt].v, (short)0, acc[mt][nt], false, false);
      }
    }
  };

  fetch(0);
  stage(0);
  __syncthreads();

  const int KT = KDIM / BK;
#pragma unroll 1
  for (int kt = 0; kt < KT; ++kt) {
    const int cur = kt & 1;
    if (kt + 1 < KT) fetch(kt + 1);
    compute(cur);
    if (kt + 1 < KT) stage(cur ^ 1);
    __syncthreads();
  }

  const int mhi = (lane >> 4) << 3;
#pragma unroll
  for (int nt = 0; nt < 2; ++nt) {
    const int n  = n0 + bn + nt * 16 + l15;
    const float bv = bias[n];
#pragma unroll
    for (int mt = 0; mt < 4; ++mt) {
#pragma unroll
      for (int r = 0; r < 8; ++r) {
        const int m = m0 + am + mt * 16 + mhi + r;
        out[(size_t)m * NDIM + n] = acc[mt][nt][r] + bv;
      }
    }
  }
}

// ---------------------------------------------------------------------------
extern "C" void kernel_launch(void* const* d_in, const int* in_sizes, int n_in,
                              void* d_out, int out_size, void* d_ws, size_t ws_size,
                              hipStream_t stream) {
  const float* x    = (const float*)d_in[0];
  const float* w    = (const float*)d_in[1];
  const float* bias = (const float*)d_in[2];
  float* out = (float*)d_out;

  const int M = in_sizes[0] / KDIM;       // 4*2048 = 8192 rows

  if (ws_size >= WS_NEEDED) {
    unsigned short* q  = (unsigned short*)((char*)d_ws + WS_Q_OFF);
    unsigned short* xh = (unsigned short*)((char*)d_ws + WS_XH_OFF);
    unsigned short* xl = (unsigned short*)((char*)d_ws + WS_XL_OFF);

    const int qblocks = (NDIM * KDIM / 8) / 256;          // 8192
    const int sblocks = (int)(((size_t)M * KDIM / 8) / 256);  // 16384
    prep_quant<<<qblocks, 256, 0, stream>>>(w, q);
    prep_split<<<sblocks, 256, 0, stream>>>(x, xh, xl);

    dim3 grid(NDIM / BN, M / BM);          // (32, 64)
    gemm_bf16x2_wmma<<<grid, dim3(256), 0, stream>>>(xh, xl, q, bias, out);
  } else {
    dim3 grid(NDIM / BN, M / BM);
    ternary_linear_fused<<<grid, dim3(256), 0, stream>>>(x, w, bias, out);
  }
}
